// QRNN_15229954032278
// MI455X (gfx1250) — matrix operand order repaired and checked by reading, not compiled, not run on hardware
//
#include <hip/hip_runtime.h>
#include <math.h>

// Problem constants
#define S_LEN   2048
#define B_SZ    32
#define D_IN    8
#define H_DIM   1024
#define Q_SZ    256

// Kernel config
#define NWG         16
#define NTHREADS    256
#define COLS_PER_WG 64
#define KSTRIDE     1056   // 1024 K + 32 tail (8 wx rows + 24 zeros)
#define SM_W_SHORTS (COLS_PER_WG * KSTRIDE)   // 67584 shorts = 135168 B
#define SM_A_SHORTS (B_SZ * KSTRIDE)          // 33792 shorts =  67584 B
#define SMEM_BYTES  ((SM_W_SHORTS + SM_A_SHORTS) * 2 + NTHREADS * 4)

typedef __attribute__((ext_vector_type(16))) __bf16 v16bf;
typedef __attribute__((ext_vector_type(8)))  __bf16 v8bf;
typedef __attribute__((ext_vector_type(8)))  float  v8f;

__device__ __forceinline__ unsigned short f2bf(float f) {
    union { float f; unsigned u; } v; v.f = f;
    unsigned u = v.u + 0x7FFFu + ((v.u >> 16) & 1u);   // round-to-nearest-even
    return (unsigned short)(u >> 16);
}
__device__ __forceinline__ float bf2f(unsigned short u) {
    union { unsigned u; float f; } v; v.u = ((unsigned)u) << 16; return v.f;
}

// W[k][n] of the Hamilton-product block matrix.
__device__ __forceinline__ float quatW(int k, int n,
                                       const float* r, const float* i,
                                       const float* j, const float* kq) {
    int br = k >> 8, bc = n >> 8;
    int idx = (k & 255) * Q_SZ + (n & 255);
    const float* m = r; float s = 1.0f;
    switch (bc * 4 + br) {
        case 0:  m = r;  s =  1.f; break;  // col_r = [ r, -i, -j, -k ]
        case 1:  m = i;  s = -1.f; break;
        case 2:  m = j;  s = -1.f; break;
        case 3:  m = kq; s = -1.f; break;
        case 4:  m = i;  s =  1.f; break;  // col_i = [ i,  r, -k,  j ]
        case 5:  m = r;  s =  1.f; break;
        case 6:  m = kq; s = -1.f; break;
        case 7:  m = j;  s =  1.f; break;
        case 8:  m = j;  s =  1.f; break;  // col_j = [ j,  k,  r, -i ]
        case 9:  m = kq; s =  1.f; break;
        case 10: m = r;  s =  1.f; break;
        case 11: m = i;  s = -1.f; break;
        case 12: m = kq; s =  1.f; break;  // col_k = [ k, -j,  i,  r ]
        case 13: m = j;  s = -1.f; break;
        case 14: m = i;  s =  1.f; break;
        case 15: m = r;  s =  1.f; break;
    }
    return s * m[idx];
}

__device__ __forceinline__ void grid_sync(unsigned* cnt, unsigned* gen) {
    __threadfence();
    __syncthreads();
    if (threadIdx.x == 0) {
        unsigned g = __hip_atomic_load(gen, __ATOMIC_RELAXED, __HIP_MEMORY_SCOPE_AGENT);
        unsigned a = __hip_atomic_fetch_add(cnt, 1u, __ATOMIC_ACQ_REL, __HIP_MEMORY_SCOPE_AGENT);
        if (a == NWG - 1) {
            __hip_atomic_store(cnt, 0u, __ATOMIC_RELAXED, __HIP_MEMORY_SCOPE_AGENT);
            __hip_atomic_fetch_add(gen, 1u, __ATOMIC_RELEASE, __HIP_MEMORY_SCOPE_AGENT);
        } else {
            while (__hip_atomic_load(gen, __ATOMIC_ACQUIRE, __HIP_MEMORY_SCOPE_AGENT) == g)
                __builtin_amdgcn_s_sleep(8);
        }
    }
    __syncthreads();
}

__device__ __forceinline__ v16bf load_afrag(const unsigned short* pA, int k) {
    v8bf lo = *(const v8bf*)(pA + k);        // K: k+off    .. +7
    v8bf hi = *(const v8bf*)(pA + k + 16);   // K: k+off+16 .. +7
    return __builtin_shufflevector(lo, hi, 0,1,2,3,4,5,6,7,8,9,10,11,12,13,14,15);
}

__global__ void qrnn_init(unsigned* hbits, unsigned* syncp) {
    int idx = blockIdx.x * blockDim.x + threadIdx.x;
    if (idx < (2 * B_SZ * H_DIM) / 2) hbits[idx] = 0u;   // 2 buffers of bf16 h
    if (idx < 2) syncp[idx] = 0u;
}

__global__ void __launch_bounds__(NTHREADS, 1)
qrnn_persistent(const float* __restrict__ x,     const float* __restrict__ wx_w,
                const float* __restrict__ wx_b,  const float* __restrict__ r_w,
                const float* __restrict__ i_w,   const float* __restrict__ j_w,
                const float* __restrict__ k_w,   const float* __restrict__ uh_b,
                const float* __restrict__ fco_w, const float* __restrict__ fco_b,
                float* __restrict__ out, unsigned short* __restrict__ hbuf,
                unsigned* __restrict__ syncp) {
    extern __shared__ char smem_raw[];
    unsigned short* smW  = (unsigned short*)smem_raw;          // [col][K] bf16, K-stride 1056
    unsigned short* smA  = smW + SM_W_SHORTS;                  // [row][K] bf16
    float*          smRd = (float*)(smA + SM_A_SHORTS);        // 256 floats, readout reduce

    const int tid   = threadIdx.x;
    const int wg    = blockIdx.x;
    const int colbase = wg * COLS_PER_WG;
    const int wave  = tid >> 5, lane = tid & 31;
    const int mt    = wave & 1;         // M tile (rows 0-15 / 16-31)
    const int nt    = wave >> 1;        // N tile within the 64-col slice
    const int lhal  = lane >> 4;        // lane half (A/B K-group selector)
    const int col_l = nt * 16 + (lane & 15);
    const int gcol  = colbase + col_l;
    const int arow  = mt * 16 + (lane & 15);

    // ---- one-time: build W slice (+ wx tail) in LDS as bf16, [col][K] ----
    for (int idx = tid; idx < SM_W_SHORTS; idx += NTHREADS) {
        int c = idx / KSTRIDE, K = idx % KSTRIDE;
        int gc = colbase + c;
        float w;
        if (K < H_DIM)              w = quatW(K, gc, r_w, i_w, j_w, k_w);
        else if (K < H_DIM + D_IN)  w = wx_w[(K - H_DIM) * H_DIM + gc];
        else                        w = 0.0f;
        smW[c * KSTRIDE + K] = f2bf(w);
    }
    // zero A tail region once (x occupies first 8 of 32 tail slots each step)
    for (int idx = tid; idx < B_SZ * 32; idx += NTHREADS) {
        int rr = idx >> 5, e = idx & 31;
        smA[rr * KSTRIDE + H_DIM + e] = 0;
    }
    const float uhb = uh_b[gcol];
    const float wxb = wx_b[gcol];
    __syncthreads();

    unsigned short* hcur = hbuf;                      // bf16, [row][1024]
    unsigned short* hnxt = hbuf + B_SZ * H_DIM;

    // LDS byte offset of the A panel (dynamic LDS starts at offset 0: no static LDS here)
    const unsigned smA_byte_base = (unsigned)(SM_W_SHORTS * 2);

    for (int t = 0; t < 2 * S_LEN; ++t) {
        const bool phase1 = (t < S_LEN);

        // ---- stage A panel: bf16 h (global) -> LDS via async copy engine ----
        {
            // 64 KB total: 4096 x 16-byte chunks, 16 per thread
            #pragma unroll
            for (int i2 = 0; i2 < 16; ++i2) {
                int ci   = tid + (i2 << 8);            // chunk id 0..4095
                int row  = ci >> 7;                    // 128 chunks per 2048-byte row
                int boff = (ci & 127) << 4;            // byte offset within row
                unsigned voff = (unsigned)((row << 11) + boff);                  // global byte off
                unsigned ldso = smA_byte_base + (unsigned)(row * (KSTRIDE * 2) + boff);
                asm volatile("global_load_async_to_lds_b128 %0, %1, %2 offset:0"
                             :: "v"(ldso), "v"(voff), "s"(hcur)
                             : "memory");
            }
            if (phase1) {
                int b = tid >> 3, d = tid & 7;
                smA[b * KSTRIDE + H_DIM + d] =
                    f2bf(x[((size_t)t * B_SZ + b) * D_IN + d]);
            }
            asm volatile("s_wait_asynccnt 0x0" ::: "memory");
        }
        __syncthreads();

        // ---- GEMM: 16x16 tile per wave; two independent WMMA chains over K ----
        v8f acc0 = {}, acc1 = {};
        const unsigned short* pA = smA + arow * KSTRIDE + lhal * 8;
        const unsigned short* pB = smW + col_l * KSTRIDE + lhal * 8;
        for (int k = 0; k < H_DIM; k += 64) {
            v16bf a0 = load_afrag(pA, k);
            v16bf b0 = load_afrag(pB, k);
            acc0 = __builtin_amdgcn_wmma_f32_16x16x32_bf16(
                       false, a0, false, b0, (short)0, acc0, false, false);
            v16bf a1 = load_afrag(pA, k + 32);
            v16bf b1 = load_afrag(pB, k + 32);
            acc1 = __builtin_amdgcn_wmma_f32_16x16x32_bf16(
                       false, a1, false, b1, (short)0, acc1, false, false);
        }
        if (phase1) {   // wx tail chunk (x[t] @ wx_w folded as K = 1024..1055)
            v16bf a2 = load_afrag(pA, H_DIM);
            v16bf b2 = load_afrag(pB, H_DIM);
            acc0 = __builtin_amdgcn_wmma_f32_16x16x32_bf16(
                       false, a2, false, b2, (short)0, acc0, false, false);
        }

        // ---- epilogue: + uh_b (+ wx_b in phase 1), store h_next (bf16) ----
        const int rbase = mt * 16 + lhal * 8;
        #pragma unroll
        for (int v = 0; v < 8; ++v) {
            float val = acc0[v] + acc1[v] + uhb;
            if (phase1) val += wxb;
            hnxt[(rbase + v) * H_DIM + gcol] = f2bf(val);
        }

        grid_sync(syncp, syncp + 1);   // full h_next now visible device-wide

        // ---- phase 2: sigmoid readout o[t-S] = sigmoid(h_new @ fco_w + b) ----
        if (!phase1) {
            const int tau  = t - S_LEN;
            const int pr   = wg * 16 + (tid >> 4);   // (b,c) pair: 256 total
            const int b    = pr >> 3, c = pr & 7;
            const int part = tid & 15;
            const unsigned short* hp = hnxt + b * H_DIM + part * 64;
            const float* wp = fco_w + (size_t)part * 64 * 8 + c;
            float s = 0.0f;
            #pragma unroll 8
            for (int k2 = 0; k2 < 64; ++k2) s += bf2f(hp[k2]) * wp[k2 * 8];
            smRd[tid] = s;
            __syncthreads();
            if (part == 0) {
                float tot = 0.0f;
                #pragma unroll
                for (int u = 0; u < 16; ++u) tot += smRd[(tid & ~15) + u];
                tot += fco_b[c];
                out[((size_t)tau * B_SZ + b) * D_IN + c] =
                    1.0f / (1.0f + __expf(-tot));
            }
            __syncthreads();
        }

        unsigned short* tmp = hcur; hcur = hnxt; hnxt = tmp;
    }
}

extern "C" void kernel_launch(void* const* d_in, const int* in_sizes, int n_in,
                              void* d_out, int out_size, void* d_ws, size_t ws_size,
                              hipStream_t stream) {
    const float* x     = (const float*)d_in[0];
    const float* wx_w  = (const float*)d_in[1];
    const float* wx_b  = (const float*)d_in[2];
    const float* r_w   = (const float*)d_in[3];
    const float* i_w   = (const float*)d_in[4];
    const float* j_w   = (const float*)d_in[5];
    const float* k_w   = (const float*)d_in[6];
    const float* uh_b  = (const float*)d_in[7];
    const float* fco_w = (const float*)d_in[8];
    const float* fco_b = (const float*)d_in[9];

    unsigned short* hbuf = (unsigned short*)d_ws;     // 2 x (32x1024) bf16
    unsigned* syncp = (unsigned*)((char*)d_ws + 2 * B_SZ * H_DIM * sizeof(unsigned short));

    (void)in_sizes; (void)n_in; (void)out_size; (void)ws_size;

    hipFuncSetAttribute((const void*)qrnn_persistent,
                        hipFuncAttributeMaxDynamicSharedMemorySize, SMEM_BYTES);

    qrnn_init<<<128, 256, 0, stream>>>((unsigned*)hbuf, syncp);
    qrnn_persistent<<<NWG, NTHREADS, SMEM_BYTES, stream>>>(
        x, wx_w, wx_b, r_w, i_w, j_w, k_w, uh_b, fco_w, fco_b,
        (float*)d_out, hbuf, syncp);
}
